// PVDSA_Res_11673721111017
// MI455X (gfx1250) — compile-verified, matrix operand order checked by reading, not run on hardware
//
#include <hip/hip_runtime.h>
#include <math.h>

// Problem constants (match reference)
#define BB  2
#define CC  64
#define NN  16384
#define KK  16
#define RR  16
#define RRR 4096   // R^3

typedef __attribute__((ext_vector_type(16))) _Float16 v16h;
typedef __attribute__((ext_vector_type(8)))  float    v8f;

// ---------------------------------------------------------------------------
// Kernel 1: per-batch xyz mean over N  (mean[b*3+d])
// ---------------------------------------------------------------------------
__global__ __launch_bounds__(256) void k_mean(const float* __restrict__ xyz,
                                              float* __restrict__ mean) {
    const int b = blockIdx.x;
    __shared__ float red[3][256];
    float s0 = 0.f, s1 = 0.f, s2 = 0.f;
    for (int n = threadIdx.x; n < NN; n += 256) {
        s0 += xyz[(b * 3 + 0) * NN + n];
        s1 += xyz[(b * 3 + 1) * NN + n];
        s2 += xyz[(b * 3 + 2) * NN + n];
    }
    red[0][threadIdx.x] = s0; red[1][threadIdx.x] = s1; red[2][threadIdx.x] = s2;
    __syncthreads();
    for (int off = 128; off > 0; off >>= 1) {
        if (threadIdx.x < (unsigned)off) {
            red[0][threadIdx.x] += red[0][threadIdx.x + off];
            red[1][threadIdx.x] += red[1][threadIdx.x + off];
            red[2][threadIdx.x] += red[2][threadIdx.x + off];
        }
        __syncthreads();
    }
    if (threadIdx.x < 3) mean[b * 3 + threadIdx.x] = red[threadIdx.x][0] * (1.0f / NN);
}

// ---------------------------------------------------------------------------
// Kernel 2: voxel scatter (segment sums + counts).  gsum layout [b][c][voxel]
// ---------------------------------------------------------------------------
__global__ __launch_bounds__(256) void k_vox(const float* __restrict__ x,
                                             const float* __restrict__ xyz,
                                             const float* __restrict__ mean,
                                             float* __restrict__ gsum,
                                             float* __restrict__ gcnt) {
    const int gid = blockIdx.x * 256 + threadIdx.x;   // 0..B*N-1 exact
    const int b = gid / NN, n = gid % NN;
    int vi[3];
    #pragma unroll
    for (int d = 0; d < 3; ++d) {
        float nc = (xyz[(b * 3 + d) * NN + n] - mean[b * 3 + d] + 1.0f) * 0.5f * (float)RR;
        nc = fminf(fmaxf(nc, 0.0f), (float)RR - 1.0f);
        vi[d] = (int)rintf(nc);   // round-half-even, matches jnp.round
    }
    const int flat = (vi[0] * RR + vi[1]) * RR + vi[2];
    atomicAdd(&gcnt[b * RRR + flat], 1.0f);
    for (int c = 0; c < CC; ++c)
        atomicAdd(&gsum[(b * CC + c) * RRR + flat], x[(b * CC + c) * NN + n]);
}

// ---------------------------------------------------------------------------
// Kernel 3: grid mean + depthwise conv3a + BN + lrelu + conv3b + BN + lrelu
// One block per (b,c).  Whole 16^3 slice lives in LDS (2 x 16 KB).
// ---------------------------------------------------------------------------
__global__ __launch_bounds__(256) void k_conv(
    const float* __restrict__ gsum, const float* __restrict__ gcnt,
    const float* __restrict__ w3a, const float* __restrict__ b3a,
    const float* __restrict__ g3a, const float* __restrict__ be3a,
    const float* __restrict__ m3a, const float* __restrict__ v3a,
    const float* __restrict__ w3b, const float* __restrict__ b3b,
    const float* __restrict__ g3b, const float* __restrict__ be3b,
    const float* __restrict__ m3b, const float* __restrict__ v3b,
    float* __restrict__ gridf) {
    const int b = blockIdx.x >> 6;
    const int c = blockIdx.x & 63;
    __shared__ float g0[RRR];
    __shared__ float g1[RRR];

    for (int v = threadIdx.x; v < RRR; v += 256) {
        float cnt = gcnt[b * RRR + v];
        g0[v] = gsum[(b * CC + c) * RRR + v] / fmaxf(cnt, 1.0f);
    }
    __syncthreads();

    float wa[27], wb[27];
    #pragma unroll
    for (int i = 0; i < 27; ++i) { wa[i] = w3a[c * 27 + i]; wb[i] = w3b[c * 27 + i]; }
    const float sA = g3a[c] * rsqrtf(v3a[c] + 1e-4f), mA = m3a[c], bnA = be3a[c], biasA = b3a[c];
    const float sB = g3b[c] * rsqrtf(v3b[c] + 1e-4f), mB = m3b[c], bnB = be3b[c], biasB = b3b[c];

    for (int v = threadIdx.x; v < RRR; v += 256) {
        const int d = v >> 8, h = (v >> 4) & 15, w = v & 15;
        float s = biasA;
        #pragma unroll
        for (int kd = 0; kd < 3; ++kd) {
            const int dd = d + kd - 1;
            if ((unsigned)dd < 16u) {
                #pragma unroll
                for (int kh = 0; kh < 3; ++kh) {
                    const int hh = h + kh - 1;
                    if ((unsigned)hh < 16u) {
                        #pragma unroll
                        for (int kw = 0; kw < 3; ++kw) {
                            const int ww = w + kw - 1;
                            if ((unsigned)ww < 16u)
                                s += g0[(dd << 8) + (hh << 4) + ww] * wa[kd * 9 + kh * 3 + kw];
                        }
                    }
                }
            }
        }
        s = (s - mA) * sA + bnA;
        g1[v] = (s > 0.f) ? s : 0.1f * s;
    }
    __syncthreads();

    for (int v = threadIdx.x; v < RRR; v += 256) {
        const int d = v >> 8, h = (v >> 4) & 15, w = v & 15;
        float s = biasB;
        #pragma unroll
        for (int kd = 0; kd < 3; ++kd) {
            const int dd = d + kd - 1;
            if ((unsigned)dd < 16u) {
                #pragma unroll
                for (int kh = 0; kh < 3; ++kh) {
                    const int hh = h + kh - 1;
                    if ((unsigned)hh < 16u) {
                        #pragma unroll
                        for (int kw = 0; kw < 3; ++kw) {
                            const int ww = w + kw - 1;
                            if ((unsigned)ww < 16u)
                                s += g1[(dd << 8) + (hh << 4) + ww] * wb[kd * 9 + kh * 3 + kw];
                        }
                    }
                }
            }
        }
        s = (s - mB) * sB + bnB;
        gridf[(b * CC + c) * RRR + v] = (s > 0.f) ? s : 0.1f * s;
    }
}

// ---------------------------------------------------------------------------
// Kernel 4: fused trilinear devox + point branch (WMMA einsum) + combine.
// 1 wave (32 lanes) = 1 point; 8 waves per block; grid = B*N/8 blocks exact.
// t(3x16) = wp1(3x64, zero-padded to 16 rows) x df(64x16) via two
// v_wmma_f32_16x16x32_f16 (K = 0..31, 32..63).
// ---------------------------------------------------------------------------
__global__ __launch_bounds__(256) void k_fused(
    const float* __restrict__ x, const float* __restrict__ xyz,
    const int* __restrict__ idx, const float* __restrict__ wp1,
    const float* __restrict__ gp, const float* __restrict__ bp,
    const float* __restrict__ mp, const float* __restrict__ vp,
    const float* __restrict__ wp2, const float* __restrict__ gt,
    const float* __restrict__ bt, const float* __restrict__ mt,
    const float* __restrict__ vt, const float* __restrict__ gridf,
    const float* __restrict__ mean, float* __restrict__ out) {
    __shared__ float xjs[8][CC * KK];   // gathered neighbor features [c][k]
    __shared__ float xcs[8][CC];        // center features
    __shared__ float dpl[8][6 * KK];    // dp channels [6][k]

    const int wv   = threadIdx.x >> 5;
    const int lane = threadIdx.x & 31;
    const int p    = blockIdx.x * 8 + wv;     // global point id, always valid
    const int b    = p / NN, n = p % NN;
    const int kn   = lane & 15;               // neighbor column owned by lane
    const int half = lane >> 4;

    // Warm L2/WGP$ for the devox grid lines of this batch (gfx1250 prefetch).
    __builtin_prefetch(&gridf[(b * CC + lane) * RRR], 0, 0);

    const int j = idx[(b * NN + n) * KK + kn];

    // ---- stage: center features + neighbor gather (each value loaded once) ----
    xcs[wv][lane]      = x[(b * CC + lane) * NN + n];
    xcs[wv][lane + 32] = x[(b * CC + lane + 32) * NN + n];
    {
        const int cbase = half * 32;
        #pragma unroll 4
        for (int c = 0; c < 32; ++c)
            xjs[wv][(cbase + c) * KK + kn] = x[(b * CC + cbase + c) * NN + j];
    }
    __syncthreads();

    // ---- WMMA operand packing per CDNA5 16-bit 16x32 A/B lane layout ----
    // lane 0-15 : element e<8 -> K=e,    e>=8 -> K=e+8
    // lane 16-31: element e<8 -> K=e+8,  e>=8 -> K=e+16
    v16h a0, a1, b0, b1;
    const int m = lane & 15;  // A-matrix row (only rows 0..2 are nonzero wp1)
    #pragma unroll
    for (int e = 0; e < 16; ++e) {
        const int kk = e + ((e < 8) ? (8 * half) : (8 + 8 * half));
        a0[e] = (_Float16)((m < 3) ? wp1[m * CC + kk]      : 0.0f);
        a1[e] = (_Float16)((m < 3) ? wp1[m * CC + kk + 32] : 0.0f);
        b0[e] = (_Float16)(xjs[wv][kk * KK + kn]        - xcs[wv][kk]);
        b1[e] = (_Float16)(xjs[wv][(kk + 32) * KK + kn] - xcs[wv][kk + 32]);
    }
    v8f acc = {};
    acc = __builtin_amdgcn_wmma_f32_16x16x32_f16(false, a0, false, b0,
                                                 (short)0, acc, false, false);
    acc = __builtin_amdgcn_wmma_f32_16x16x32_f16(false, a1, false, b1,
                                                 (short)0, acc, false, false);

    // ---- finish point-branch dp channels (lanes 0-15, one per neighbor) ----
    if (lane < 16) {
        // D layout: lanes 0-15 hold column N=lane; VGPR r holds row M=r.
        float tr[3] = { acc[0], acc[1], acc[2] };
        float tt[3];
        #pragma unroll
        for (int o = 0; o < 3; ++o) {
            float r = (tr[o] - mp[o]) * (gp[o] * rsqrtf(vp[o] + 1e-5f)) + bp[o];
            tt[o] = fmaxf(r, 0.0f);
        }
        #pragma unroll
        for (int o = 0; o < 3; ++o)
            dpl[wv][(3 + o) * KK + kn] =
                wp2[o * 3 + 0] * tt[0] + wp2[o * 3 + 1] * tt[1] + wp2[o * 3 + 2] * tt[2];
        #pragma unroll
        for (int d = 0; d < 3; ++d)
            dpl[wv][d * KK + kn] = xyz[(b * 3 + d) * NN + j] - xyz[(b * 3 + d) * NN + n];
    }
    __syncthreads();

    // ---- trilinear devox coefficients (cheap, computed per lane) ----
    float fr[3]; int c0[3], c1[3];
    #pragma unroll
    for (int d = 0; d < 3; ++d) {
        float v = (xyz[(b * 3 + d) * NN + n] - mean[b * 3 + d] + 1.0f) * 0.5f * (float)RR;
        v = fminf(fmaxf(v, 0.0f), (float)RR - 1.0f);
        c0[d] = (int)floorf(v);
        c1[d] = min(c0[d] + 1, RR - 1);
        fr[d] = v - (float)c0[d];
    }

    // ---- per-channel: devox + weighted max-pool; lane owns c and c+32 ----
    #pragma unroll
    for (int t2 = 0; t2 < 2; ++t2) {
        const int c = lane + 32 * t2;
        const float* gf = &gridf[(b * CC + c) * RRR];
        float pv = 0.0f;
        #pragma unroll
        for (int tap = 0; tap < 8; ++tap) {
            const int dx = tap >> 2, dy = (tap >> 1) & 1, dz = tap & 1;
            const int ix = dx ? c1[0] : c0[0];
            const int iy = dy ? c1[1] : c0[1];
            const int iz = dz ? c1[2] : c0[2];
            const float w = (dx ? fr[0] : 1.f - fr[0]) *
                            (dy ? fr[1] : 1.f - fr[1]) *
                            (dz ? fr[2] : 1.f - fr[2]);
            pv += gf[(ix * RR + iy) * RR + iz] * w;
        }
        const float sc = gt[c] * rsqrtf(vt[c] + 1e-5f);
        const float mc = mt[c], bc = bt[c];
        const int rid = (c < 50) ? (c / 10) : 5;   // _RIDX partition map
        float pmax = 0.0f;                         // relu'd values are >= 0
        #pragma unroll 4
        for (int k = 0; k < KK; ++k) {
            float v = xjs[wv][c * KK + k] * dpl[wv][rid * KK + k];
            v = (v - mc) * sc + bc;
            v = fmaxf(v, 0.0f);
            pmax = fmaxf(pmax, v);
        }
        out[(b * CC + c) * NN + n] = pv + pmax;
    }
}

// ---------------------------------------------------------------------------
// Host launcher
// ---------------------------------------------------------------------------
extern "C" void kernel_launch(void* const* d_in, const int* in_sizes, int n_in,
                              void* d_out, int out_size, void* d_ws, size_t ws_size,
                              hipStream_t stream) {
    const float* x    = (const float*)d_in[0];
    const float* xyz  = (const float*)d_in[1];
    const int*   idx  = (const int*)  d_in[2];
    const float* w3a  = (const float*)d_in[3];
    const float* b3a  = (const float*)d_in[4];
    const float* g3a  = (const float*)d_in[5];
    const float* be3a = (const float*)d_in[6];
    const float* m3a  = (const float*)d_in[7];
    const float* v3a  = (const float*)d_in[8];
    const float* w3b  = (const float*)d_in[9];
    const float* b3b  = (const float*)d_in[10];
    const float* g3b  = (const float*)d_in[11];
    const float* be3b = (const float*)d_in[12];
    const float* m3b  = (const float*)d_in[13];
    const float* v3b  = (const float*)d_in[14];
    const float* wp1  = (const float*)d_in[15];
    const float* gp   = (const float*)d_in[16];
    const float* bp   = (const float*)d_in[17];
    const float* mp   = (const float*)d_in[18];
    const float* vp   = (const float*)d_in[19];
    const float* wp2  = (const float*)d_in[20];
    const float* gt   = (const float*)d_in[21];
    const float* bt   = (const float*)d_in[22];
    const float* mt   = (const float*)d_in[23];
    const float* vt   = (const float*)d_in[24];
    float* out = (float*)d_out;

    // Workspace layout
    char* ws = (char*)d_ws;
    const size_t mean_bytes = 256;                          // 6 floats, padded
    const size_t cnt_bytes  = (size_t)BB * RRR * 4;         // 32 KB
    const size_t sum_bytes  = (size_t)BB * CC * RRR * 4;    // 2 MB
    float* mean  = (float*)(ws);
    float* gcnt  = (float*)(ws + mean_bytes);
    float* gsum  = (float*)(ws + mean_bytes + cnt_bytes);
    float* gridf = (float*)(ws + mean_bytes + cnt_bytes + sum_bytes);

    // Zero the scatter accumulators (contiguous: gcnt then gsum).
    hipMemsetAsync(gcnt, 0, cnt_bytes + sum_bytes, stream);

    k_mean<<<BB, 256, 0, stream>>>(xyz, mean);
    k_vox <<<(BB * NN) / 256, 256, 0, stream>>>(x, xyz, mean, gsum, gcnt);
    k_conv<<<BB * CC, 256, 0, stream>>>(gsum, gcnt,
                                        w3a, b3a, g3a, be3a, m3a, v3a,
                                        w3b, b3b, g3b, be3b, m3b, v3b, gridf);
    k_fused<<<(BB * NN) / 8, 256, 0, stream>>>(x, xyz, idx, wp1, gp, bp, mp, vp,
                                               wp2, gt, bt, mt, vt, gridf, mean, out);
}
